// GAT_PoseNet_2671469658733
// MI455X (gfx1250) — matrix operand order repaired and checked by reading, not compiled, not run on hardware
//
#include <hip/hip_runtime.h>
#include <hip/hip_bf16.h>

typedef __attribute__((ext_vector_type(16))) _Float16 v16h;
typedef __attribute__((ext_vector_type(8)))  float    v8f;

#define NN    20000
#define EE    320000
#define ETOT  (EE + NN)
#define DIN   128
#define HID   1024
#define MIDD  256
#define DOUT  64

// ------------------------------------------------- async global->LDS support

#if defined(__gfx1250__) && __has_builtin(__builtin_amdgcn_global_load_async_to_lds_b128)
#define HAVE_ASYNC_LDS 1
#else
#define HAVE_ASYNC_LDS 0
#endif

#if HAVE_ASYNC_LDS
typedef int v4i_t __attribute__((ext_vector_type(4)));
typedef __attribute__((address_space(1))) v4i_t gv4i_t;   // global (prints as __device__)
typedef __attribute__((address_space(3))) v4i_t lv4i_t;   // LDS
// 16B async copy: global -> LDS, tracked by ASYNCcnt, no VGPR staging.
__device__ __forceinline__ void async_copy16(const void* g, void* l) {
  __builtin_amdgcn_global_load_async_to_lds_b128(
      (gv4i_t*)(uintptr_t)g,
      (lv4i_t*)(unsigned)(uintptr_t)l,   // low 32 bits of generic addr == LDS offset
      0, 0);
}
__device__ __forceinline__ void async_wait0() {
#if __has_builtin(__builtin_amdgcn_s_wait_asynccnt)
  __builtin_amdgcn_s_wait_asynccnt(0);
#else
  asm volatile("s_wait_asynccnt 0x0" ::: "memory");
#endif
}
#else
__device__ __forceinline__ void async_wait0() {}
#endif

// ---------------------------------------------------------------- utilities

__global__ void zero_f32_kernel(float* p, long long n) {
  long long i = (long long)blockIdx.x * blockDim.x + threadIdx.x;
  long long st = (long long)gridDim.x * blockDim.x;
  for (; i < n; i += st) p[i] = 0.0f;
}

__global__ void cvt_f32_f16_kernel(const float* __restrict__ in, _Float16* __restrict__ out, long long n) {
  long long i = (long long)blockIdx.x * blockDim.x + threadIdx.x;
  long long st = (long long)gridDim.x * blockDim.x;
  for (; i < n; i += st) out[i] = (_Float16)in[i];
}

// W[K][N] fp32 -> Wt[N][K] f16
__global__ void cvt_T_kernel(const float* __restrict__ W, _Float16* __restrict__ Wt, int K, int N) {
  long long i = (long long)blockIdx.x * blockDim.x + threadIdx.x;
  long long n = (long long)K * N;
  long long st = (long long)gridDim.x * blockDim.x;
  for (; i < n; i += st) {
    int k = (int)(i / N), c = (int)(i % N);
    Wt[(long long)c * K + k] = (_Float16)W[i];
  }
}

// ---------------------------------------------------------------- WMMA GEMM
// C[M][N] = A[M][K] (f16, row-major) @ Bt[N][K]^T (f16, N-major == B col-major)
// block tile 128x64, 8 waves (4 along M x 2 along N), wave tile 32x32, K-step 32.
// Double-buffered LDS staging via GLOBAL_LOAD_ASYNC_TO_LDS_B128 (ASYNCcnt).

#define LDT 40   // LDS row stride in halves (80B: 16B-aligned, bank-spread)

template<bool HAS_BIAS, bool RELU, bool OUT_HALF>
__global__ __launch_bounds__(256) void gemm_wmma_kernel(
    const _Float16* __restrict__ A, const _Float16* __restrict__ Bt,
    const float* __restrict__ bias, void* __restrict__ Cout,
    int M, int N, int K)
{
  __shared__ _Float16 sA[2][128 * LDT];
  __shared__ _Float16 sB[2][64 * LDT];
  const int tid  = threadIdx.x;
  const int lane = tid & 31;
  const int wave = tid >> 5;
  const int wm = wave & 3;         // 0..3  -> 32-row strip
  const int wn = wave >> 2;        // 0..1  -> 32-col strip
  const int half = lane >> 4;      // hi/lo 16 lanes
  const int lm   = lane & 15;
  const long long bM = (long long)blockIdx.x * 128;
  const long long bN = (long long)blockIdx.y * 64;
  const bool fullM = (bM + 128 <= (long long)M);

  v8f acc[2][2] = {};

  // stage one 128x32 A tile + 64x32 B tile into LDS buffer `buf`
  auto stage = [&](int buf, int k0) {
#if HAVE_ASYNC_LDS
#pragma unroll
    for (int s = 0; s < 2; ++s) {
      int slot = tid + s * 256;
      int row = slot >> 2, c = slot & 3;
      long long gr = bM + row;
      _Float16* lp = &sA[buf][row * LDT + c * 8];
      if (fullM || gr < M) {
        async_copy16(A + gr * (long long)K + k0 + c * 8, lp);
      } else {
        *reinterpret_cast<uint4*>(lp) = make_uint4(0u, 0u, 0u, 0u);
      }
    }
    {
      int row = tid >> 2, c = tid & 3;
      async_copy16(Bt + (bN + row) * (long long)K + k0 + c * 8,
                   &sB[buf][row * LDT + c * 8]);
    }
#else
#pragma unroll
    for (int s = 0; s < 2; ++s) {
      int slot = tid + s * 256;
      int row = slot >> 2, c = slot & 3;
      long long gr = bM + row;
      uint4 v = make_uint4(0u, 0u, 0u, 0u);
      if (gr < M) v = *reinterpret_cast<const uint4*>(A + gr * (long long)K + k0 + c * 8);
      *reinterpret_cast<uint4*>(&sA[buf][row * LDT + c * 8]) = v;
    }
    {
      int row = tid >> 2, c = tid & 3;
      uint4 v = *reinterpret_cast<const uint4*>(Bt + (bN + row) * (long long)K + k0 + c * 8);
      *reinterpret_cast<uint4*>(&sB[buf][row * LDT + c * 8]) = v;
    }
#endif
  };

  const int T = K >> 5;   // K/32 steps (K is a multiple of 32)
  stage(0, 0);

  for (int t = 0; t < T; ++t) {
    const int buf = t & 1;
    async_wait0();          // this wave's copies for tile t have landed in LDS
    __syncthreads();        // ... and everyone else's
    if (t + 1 < T) stage(buf ^ 1, (t + 1) << 5);   // overlap next tile with WMMAs

    const _Float16* cA = &sA[buf][0];
    const _Float16* cB = &sB[buf][0];
    v16h af[2], bf[2];
#pragma unroll
    for (int i = 0; i < 2; ++i) {       // A fragments: rows wm*32 + i*16 + lm
      int r = wm * 32 + i * 16 + lm;
#pragma unroll
      for (int j = 0; j < 8; ++j) {
        int kk = (j < 4 ? 2 * j : 16 + 2 * (j - 4)) + 8 * half;   // ISA 16-bit A layout
        af[i][2 * j]     = cA[r * LDT + kk];
        af[i][2 * j + 1] = cA[r * LDT + kk + 1];
      }
    }
#pragma unroll
    for (int i = 0; i < 2; ++i) {       // B fragments: cols wn*32 + i*16 + lm
      int r = wn * 32 + i * 16 + lm;
#pragma unroll
      for (int j = 0; j < 8; ++j) {
        int kk = (j < 4 ? 2 * j : 16 + 2 * (j - 4)) + 8 * half;   // ISA 16-bit B layout (same k order as A)
        bf[i][2 * j]     = cB[r * LDT + kk];
        bf[i][2 * j + 1] = cB[r * LDT + kk + 1];
      }
    }
#pragma unroll
    for (int i = 0; i < 2; ++i)
#pragma unroll
      for (int j = 0; j < 2; ++j)
        acc[i][j] = __builtin_amdgcn_wmma_f32_16x16x32_f16(
            false, af[i], false, bf[j], (short)0, acc[i][j], false, false);
    __syncthreads();        // all waves done reading buf before it is re-staged
  }

  // store: C/D layout — lane -> (n = lane%16, m = 8*(lane/16) + vgpr)
  const int mh = half * 8;
#pragma unroll
  for (int i = 0; i < 2; ++i) {
#pragma unroll
    for (int j = 0; j < 2; ++j) {
      long long col = bN + wn * 32 + j * 16 + lm;
      float bv = HAS_BIAS ? bias[col] : 0.0f;
#pragma unroll
      for (int r = 0; r < 8; ++r) {
        long long row = bM + wm * 32 + i * 16 + mh + r;
        if (row < M) {
          float v = acc[i][j][r] + bv;
          if (RELU) v = fmaxf(v, 0.0f);
          if (OUT_HALF) ((_Float16*)Cout)[row * (long long)N + col] = (_Float16)v;
          else          ((float*)Cout)[row * (long long)N + col] = v;
        }
      }
    }
  }
}

// ---------------------------------------------------------------- attention

// per-node dual dot products: als = h.a_src, ald = h.a_dst (wave per node)
__global__ __launch_bounds__(256) void alphas_kernel(
    const float* __restrict__ h, const float* __restrict__ asrc, const float* __restrict__ adst,
    int rows, int F, float* __restrict__ als, float* __restrict__ ald)
{
  int wave = threadIdx.x >> 5, lane = threadIdx.x & 31;
  long long row = (long long)blockIdx.x * (blockDim.x >> 5) + wave;
  if (row >= rows) return;
  const float* hr = h + row * (long long)F;
  float s = 0.0f, d = 0.0f;
  for (int f = lane; f < F; f += 32) { float v = hr[f]; s += v * asrc[f]; d += v * adst[f]; }
#pragma unroll
  for (int o = 16; o; o >>= 1) { s += __shfl_xor(s, o, 32); d += __shfl_xor(d, o, 32); }
  if (lane == 0) { als[row] = s; ald[row] = d; }
}

__device__ __forceinline__ unsigned f2ord(float f) {
  unsigned b = __float_as_uint(f);
  return (b & 0x80000000u) ? ~b : (b | 0x80000000u);
}
__device__ __forceinline__ float ord2f(unsigned o) {
  unsigned b = (o & 0x80000000u) ? (o & 0x7fffffffu) : ~o;
  return __uint_as_float(b);
}

__device__ __forceinline__ void edge_sd(const int* ei, int e, int& s, int& d) {
  if (e < EE) { s = ei[e]; d = ei[EE + e]; } else { s = d = e - EE; }
}

__global__ void edge_max_kernel(const int* __restrict__ ei, const float* __restrict__ als,
                                const float* __restrict__ ald, unsigned* __restrict__ mord)
{
  int e = blockIdx.x * blockDim.x + threadIdx.x;
  if (e >= ETOT) return;
  int s, d; edge_sd(ei, e, s, d);
  float v = als[s] + ald[d];
  v = v > 0.0f ? v : 0.2f * v;                       // leaky_relu(0.2)
  atomicMax(&mord[d], f2ord(v));
}

__global__ void edge_expsum_kernel(const int* __restrict__ ei, const float* __restrict__ als,
                                   const float* __restrict__ ald, const unsigned* __restrict__ mord,
                                   float* __restrict__ wco, float* __restrict__ den)
{
  int e = blockIdx.x * blockDim.x + threadIdx.x;
  if (e >= ETOT) return;
  int s, d; edge_sd(ei, e, s, d);
  float v = als[s] + ald[d];
  v = v > 0.0f ? v : 0.2f * v;
  float w = expf(v - ord2f(mord[d]));
  wco[e] = w;
  atomicAdd(&den[d], w);
}

// block per edge: agg[dst] += (w/denom) * h[src]   (h + agg both fit in 192MB L2)
__global__ __launch_bounds__(256) void edge_agg_kernel(
    const int* __restrict__ ei, const float* __restrict__ h, const float* __restrict__ wco,
    const float* __restrict__ den, float* __restrict__ agg, int F)
{
  int e = blockIdx.x;
  int s, d; edge_sd(ei, e, s, d);
  float c = wco[e] / (den[d] + 1e-16f);
  const float* hs = h + (long long)s * F;
  float* od = agg + (long long)d * F;
  for (int f = threadIdx.x; f < F; f += blockDim.x)
    atomicAdd(&od[f], c * hs[f]);
}

// ---------------------------------------------------------------- batchnorm

__global__ void bn_stats_kernel(const float* __restrict__ agg, const float* __restrict__ bias,
                                int rows, int F, int rchunk,
                                float* __restrict__ csum, float* __restrict__ csq)
{
  int c = blockIdx.x * blockDim.x + threadIdx.x;
  if (c >= F) return;
  int r0 = blockIdx.y * rchunk;
  int r1 = r0 + rchunk; if (r1 > rows) r1 = rows;
  float b = bias[c], s = 0.0f, q = 0.0f;
  for (int r = r0; r < r1; ++r) { float v = agg[(long long)r * F + c] + b; s += v; q += v * v; }
  atomicAdd(&csum[c], s);
  atomicAdd(&csq[c], q);
}

// fused bias + BN + ReLU, emit f16 for the next WMMA GEMM
__global__ void bn_apply_kernel(const float* __restrict__ agg, const float* __restrict__ bias,
                                const float* __restrict__ csum, const float* __restrict__ csq,
                                const float* __restrict__ g, const float* __restrict__ be,
                                int rows, int F, _Float16* __restrict__ out16)
{
  long long i = (long long)blockIdx.x * blockDim.x + threadIdx.x;
  long long n = (long long)rows * F;
  long long st = (long long)gridDim.x * blockDim.x;
  float inv_rows = 1.0f / rows;
  for (; i < n; i += st) {
    int c = (int)(i % F);
    float mu = csum[c] * inv_rows;
    float var = csq[c] * inv_rows - mu * mu;
    float v = (agg[i] + bias[c] - mu) * rsqrtf(var + 1e-5f) * g[c] + be[c];
    out16[i] = (_Float16)fmaxf(v, 0.0f);
  }
}

// ---------------------------------------------------------------- softmax

__global__ __launch_bounds__(256) void log_softmax64_kernel(float* __restrict__ out, int rows)
{
  int wave = threadIdx.x >> 5, lane = threadIdx.x & 31;
  long long row = (long long)blockIdx.x * (blockDim.x >> 5) + wave;
  if (row >= rows) return;
  float* p = out + row * 64;
  float v0 = p[lane], v1 = p[lane + 32];
  float m = fmaxf(v0, v1);
#pragma unroll
  for (int o = 16; o; o >>= 1) m = fmaxf(m, __shfl_xor(m, o, 32));
  float s = expf(v0 - m) + expf(v1 - m);
#pragma unroll
  for (int o = 16; o; o >>= 1) s += __shfl_xor(s, o, 32);
  float ls = m + logf(s);
  p[lane] = v0 - ls;
  p[lane + 32] = v1 - ls;
}

// ---------------------------------------------------------------- host

static inline dim3 gemm_grid(int M, int N) { return dim3((M + 127) / 128, N / 64); }

extern "C" void kernel_launch(void* const* d_in, const int* in_sizes, int n_in,
                              void* d_out, int out_size, void* d_ws, size_t ws_size,
                              hipStream_t stream)
{
  const float* x     = (const float*)d_in[0];
  const int*   ei    = (const int*)  d_in[1];
  const float* W1    = (const float*)d_in[2];
  const float* as1   = (const float*)d_in[3];
  const float* ad1   = (const float*)d_in[4];
  const float* b1    = (const float*)d_in[5];
  const float* g1    = (const float*)d_in[6];
  const float* be1   = (const float*)d_in[7];
  const float* W2    = (const float*)d_in[8];
  const float* as2   = (const float*)d_in[9];
  const float* ad2   = (const float*)d_in[10];
  const float* b2    = (const float*)d_in[11];
  const float* g2    = (const float*)d_in[12];
  const float* be2   = (const float*)d_in[13];
  const float* Wl1   = (const float*)d_in[14];
  const float* bl1   = (const float*)d_in[15];
  const float* Wl2   = (const float*)d_in[16];
  const float* bl2   = (const float*)d_in[17];
  float* out = (float*)d_out;

  // workspace carve (256B aligned)
  char* w = (char*)d_ws;
  auto carve = [&](size_t bytes) { char* p = w; w += (bytes + 255) & ~(size_t)255; return p; };
  float*    h     = (float*)   carve((size_t)NN * HID * 4);
  float*    agg   = (float*)   carve((size_t)NN * HID * 4);
  _Float16* y16   = (_Float16*)carve((size_t)NN * HID * 2);   // also holds x16 before layer1
  _Float16* wt    = (_Float16*)carve((size_t)1024 * 1024 * 2);
  float*    als   = (float*)   carve((size_t)NN * 4);
  float*    ald   = (float*)   carve((size_t)NN * 4);
  unsigned* mord  = (unsigned*)carve((size_t)NN * 4);
  float*    den   = (float*)   carve((size_t)NN * 4);
  float*    wco   = (float*)   carve((size_t)ETOT * 4);
  _Float16* mid16 = (_Float16*)carve((size_t)NN * MIDD * 2);
  float*    csum  = (float*)   carve((size_t)HID * 4);
  float*    csq   = (float*)   carve((size_t)HID * 4);

  const int TB = 256;
  const int egrid = (ETOT + TB - 1) / TB;
  const int rchunk = (NN + 15) / 16;

  auto gat_layer = [&](const _Float16* a16, const float* Wsrc, int K,
                       const float* asv, const float* adv, const float* bv,
                       const float* gv, const float* bev) {
    cvt_T_kernel<<<512, TB, 0, stream>>>(Wsrc, wt, K, HID);
    gemm_wmma_kernel<false, false, false><<<gemm_grid(NN, HID), TB, 0, stream>>>(
        a16, wt, nullptr, h, NN, HID, K);
    alphas_kernel<<<(NN + 7) / 8, TB, 0, stream>>>(h, asv, adv, NN, HID, als, ald);
    zero_f32_kernel<<<2048, TB, 0, stream>>>(agg, (long long)NN * HID);
    zero_f32_kernel<<<64, TB, 0, stream>>>((float*)mord, NN);   // ord(-inf) > 0, so 0 is identity
    zero_f32_kernel<<<64, TB, 0, stream>>>(den, NN);
    edge_max_kernel<<<egrid, TB, 0, stream>>>(ei, als, ald, mord);
    edge_expsum_kernel<<<egrid, TB, 0, stream>>>(ei, als, ald, mord, wco, den);
    edge_agg_kernel<<<ETOT, TB, 0, stream>>>(ei, h, wco, den, agg, HID);
    zero_f32_kernel<<<8, TB, 0, stream>>>(csum, HID);
    zero_f32_kernel<<<8, TB, 0, stream>>>(csq, HID);
    bn_stats_kernel<<<dim3(HID / TB, 16), TB, 0, stream>>>(agg, bv, NN, HID, rchunk, csum, csq);
    bn_apply_kernel<<<2048, TB, 0, stream>>>(agg, bv, csum, csq, gv, bev, NN, HID, y16);
  };

  // layer 1: x -> f16 (staged in y16 buffer, consumed before BN overwrites it)
  cvt_f32_f16_kernel<<<1024, TB, 0, stream>>>(x, y16, (long long)NN * DIN);
  gat_layer(y16, W1, DIN, as1, ad1, b1, g1, be1);
  // layer 2
  gat_layer(y16, W2, HID, as2, ad2, b2, g2, be2);

  // MLP: relu(y @ Wl1 + bl1) -> mid16 (f16), then logits = mid @ Wl2 + bl2 -> d_out
  cvt_T_kernel<<<256, TB, 0, stream>>>(Wl1, wt, HID, MIDD);
  gemm_wmma_kernel<true, true, true><<<gemm_grid(NN, MIDD), TB, 0, stream>>>(
      y16, wt, bl1, mid16, NN, MIDD, HID);
  cvt_T_kernel<<<64, TB, 0, stream>>>(Wl2, wt, MIDD, DOUT);
  gemm_wmma_kernel<true, false, false><<<gemm_grid(NN, DOUT), TB, 0, stream>>>(
      mid16, wt, bl2, out, NN, DOUT, MIDD);

  log_softmax64_kernel<<<(NN + 7) / 8, TB, 0, stream>>>(out, NN);
}
